// MHA_27238682591490
// MI455X (gfx1250) — matrix operand order repaired and checked by reading, not compiled, not run on hardware
//
#include <hip/hip_runtime.h>
#include <hip/hip_bf16.h>

typedef __attribute__((ext_vector_type(16))) _Float16 v16h;
typedef __attribute__((ext_vector_type(8)))  float    v8f;

// Async global->LDS copy (CDNA5, ASYNCcnt). dst = LDS byte offset (low 32 bits
// of the flat shared-aperture address), src = 64-bit global address.
__device__ __forceinline__ void async_copy_b128(void* lds_dst, const void* gsrc) {
  unsigned dst = (unsigned)(size_t)lds_dst;
  asm volatile("global_load_async_to_lds_b128 %0, %1, off"
               :: "v"(dst), "v"(gsrc) : "memory");
}
__device__ __forceinline__ void async_wait0() {
  asm volatile("s_wait_asynccnt 0x0" ::: "memory");
}

// ---- WMMA fragment helpers (contiguous k order, same in A and B, wave32) ----------
// A 16x32 f16: lane l<16 -> row l, K = 0..15; lane l>=16 -> row l-16,
// K = 16..31. Storage: row-major [m][k] in LDS.
__device__ __forceinline__ v16h load_a_frag(const _Float16* lds, int stride,
                                            int row0, int k0, int lane) {
  const int ln = lane & 15, hi = lane >> 4;
  const _Float16* p = lds + (row0 + ln) * stride + k0 + hi * 16;
  v16h a;
#pragma unroll
  for (int j = 0; j < 16; ++j) a[j] = p[j];
  return a;
}

// B 32x16 f16: lane l<16 -> col l, K=0..15; lane l>=16 -> col l-16, K=16..31.
// Storage: [n][k] row-major in LDS -> contiguous 16-half reads.
__device__ __forceinline__ v16h load_b_frag(const _Float16* lds, int stride,
                                            int n0, int k0, int lane) {
  const int ln = lane & 15, hi = lane >> 4;
  const _Float16* p = lds + (n0 + ln) * stride + k0 + hi * 16;
  v16h b;
#pragma unroll
  for (int j = 0; j < 16; ++j) b[j] = p[j];
  return b;
}

__device__ __forceinline__ v8f wmma16(v16h a, v16h b, v8f c) {
  return __builtin_amdgcn_wmma_f32_16x16x32_f16(false, a, false, b, (short)0, c,
                                                false, false);
}

// ---------------------------------------------------------------------------
// Kernel 1: QKV projection. C[m,f] = sum_e inp[m,e]*qkv_w[f,e] + qkv_b[f]
// Double-buffered LDS tiles; fp32 loads batched into registers, converted to
// f16 after the WMMA block so VMEM latency hides under matrix ops.
// ---------------------------------------------------------------------------
__global__ __launch_bounds__(256) void qkv_gemm_kernel(
    const float* __restrict__ inp, const float* __restrict__ qkv_w,
    const float* __restrict__ qkv_b, _Float16* __restrict__ Qh,
    _Float16* __restrict__ Kh, _Float16* __restrict__ Vh) {
  __shared__ alignas(16) _Float16 As[2][128 * 40];  // [m][e] tiles
  __shared__ alignas(16) _Float16 Bs[2][128 * 40];  // [f][e] tiles
  const int tid = threadIdx.x, lane = tid & 31, wave = tid >> 5;
  const int mTile = blockIdx.x * 128;
  const int nTile = blockIdx.y * 128;

  const v8f zero = {0.f, 0.f, 0.f, 0.f, 0.f, 0.f, 0.f, 0.f};
  v8f acc[8];
#pragma unroll
  for (int i = 0; i < 8; ++i) acc[i] = zero;

  float4 fa[4], fb[4];
  auto load_regs = [&](int k0) {
#pragma unroll
    for (int i = 0; i < 4; ++i) {
      int idx = tid + i * 256;           // 1024 chunks of 4 floats
      int row = idx >> 3, cg = idx & 7;  // 128 rows x 8 groups
      fa[i] = *(const float4*)(inp + (size_t)(mTile + row) * 1024 + k0 + cg * 4);
      fb[i] = *(const float4*)(qkv_w + (size_t)(nTile + row) * 1024 + k0 + cg * 4);
    }
  };
  auto store_tile = [&](int buf) {
#pragma unroll
    for (int i = 0; i < 4; ++i) {
      int idx = tid + i * 256;
      int row = idx >> 3, cg = idx & 7;
      _Float16* d = &As[buf][row * 40 + cg * 4];
      d[0] = (_Float16)fa[i].x; d[1] = (_Float16)fa[i].y;
      d[2] = (_Float16)fa[i].z; d[3] = (_Float16)fa[i].w;
      _Float16* e = &Bs[buf][row * 40 + cg * 4];
      e[0] = (_Float16)fb[i].x; e[1] = (_Float16)fb[i].y;
      e[2] = (_Float16)fb[i].z; e[3] = (_Float16)fb[i].w;
    }
  };

  load_regs(0);
  store_tile(0);
  __syncthreads();

  for (int k0 = 0; k0 < 1024; k0 += 32) {
    const int cur = (k0 >> 5) & 1;
    const bool more = (k0 + 32) < 1024;
    if (more) load_regs(k0 + 32);           // issue next-tile global loads
    v16h a = load_a_frag(As[cur], 40, wave * 16, 0, lane);
#pragma unroll
    for (int ns = 0; ns < 8; ++ns) {
      v16h b = load_b_frag(Bs[cur], 40, ns * 16, 0, lane);
      acc[ns] = wmma16(a, b, acc[ns]);
    }
    if (more) store_tile(cur ^ 1);          // convert + fill other buffer
    __syncthreads();
  }

  const int ln = lane & 15, hi = lane >> 4;
  const int m_base = mTile + wave * 16;
#pragma unroll
  for (int ns = 0; ns < 8; ++ns) {
    int f = nTile + ns * 16 + ln;
    float bias = qkv_b[f];
    int which = f >> 10, rem = f & 1023;
    int h = rem >> 6, d = rem & 63;
    _Float16* dst = (which == 0) ? Qh : (which == 1) ? Kh : Vh;
#pragma unroll
    for (int r = 0; r < 8; ++r) {
      int m = m_base + r + 8 * hi;          // C/D layout: VGPR r, hi -> row+8
      int b = m >> 11, s = m & 2047;
      dst[((((size_t)b * 16 + h) * 2048) + s) * 64 + d] =
          (_Float16)(acc[ns][r] + bias);
    }
  }
}

// ---------------------------------------------------------------------------
// Kernel 2: flash attention per (b,h), 128 query rows per block, 64-key tiles.
// Q/K tiles staged with async global->LDS copies (ASYNCcnt path).
// ---------------------------------------------------------------------------
__global__ __launch_bounds__(256) void attn_kernel(
    const _Float16* __restrict__ Qh, const _Float16* __restrict__ Kh,
    const _Float16* __restrict__ Vh, _Float16* __restrict__ ctx) {
  __shared__ alignas(16) _Float16 Qs[128 * 80];  // [q][d]
  __shared__ alignas(16) _Float16 Ks[64 * 80];   // [key][d] (B-frag for Q*K^T)
  __shared__ alignas(16) _Float16 Vt[64 * 80];   // [d][key] (B-frag for P*V)
  __shared__ alignas(16) _Float16 Ps[128 * 80];  // [q][key] probabilities
  const int tid = threadIdx.x, lane = tid & 31, wave = tid >> 5;
  const int ln = lane & 15, hi = lane >> 4;
  const int bh = blockIdx.x;            // b*16 + h
  const int qTile = blockIdx.y * 128;
  const _Float16* Qg = Qh + (size_t)bh * 2048 * 64;
  const _Float16* Kg = Kh + (size_t)bh * 2048 * 64;
  const _Float16* Vg = Vh + (size_t)bh * 2048 * 64;

#pragma unroll
  for (int i = 0; i < 4; ++i) {         // Q tile: async copy, 16B chunks
    int idx = tid + i * 256;
    int row = idx >> 3, cg = idx & 7;
    async_copy_b128(Qs + row * 80 + cg * 8,
                    Qg + (size_t)(qTile + row) * 64 + cg * 8);
  }

  const v8f zero = {0.f, 0.f, 0.f, 0.f, 0.f, 0.f, 0.f, 0.f};
  v8f o[4];
#pragma unroll
  for (int i = 0; i < 4; ++i) o[i] = zero;
  float mrow[8], lrow[8];
#pragma unroll
  for (int r = 0; r < 8; ++r) { mrow[r] = -__builtin_huge_valf(); lrow[r] = 0.f; }

  for (int kt = 0; kt < 2048; kt += 64) {
#pragma unroll
    for (int i = 0; i < 2; ++i) {       // K tile: async copy
      int idx = tid + i * 256;
      int row = idx >> 2, cg = idx & 3;
      async_copy_b128(Ks + row * 80 + cg * 8,
                      Kg + (size_t)(kt + row) * 64 + cg * 8);
    }
    uint4 vv[2];                        // V tile: batched loads, then transpose
#pragma unroll
    for (int i = 0; i < 2; ++i) {
      int idx = tid + i * 256;
      int row = idx >> 2, cg = idx & 3;
      vv[i] = *(const uint4*)(Vg + (size_t)(kt + row) * 64 + cg * 8);
    }
#pragma unroll
    for (int i = 0; i < 2; ++i) {
      int idx = tid + i * 256;
      int row = idx >> 2, cg = idx & 3;
      const _Float16* vp = (const _Float16*)&vv[i];
#pragma unroll
      for (int j = 0; j < 8; ++j) Vt[(cg * 8 + j) * 80 + row] = vp[j];
    }
    async_wait0();                      // Q (iter 0) + K async ops complete
    __syncthreads();

    // scores: S[q,key] = sum_d Q[q,d]*K[key,d];  B[k=d][n=key] <- Ks[key][d]
    v8f s[4];
#pragma unroll
    for (int i = 0; i < 4; ++i) s[i] = zero;
#pragma unroll
    for (int ks = 0; ks < 64; ks += 32) {
      v16h a = load_a_frag(Qs, 80, wave * 16, ks, lane);
#pragma unroll
      for (int nsub = 0; nsub < 4; ++nsub) {
        v16h b = load_b_frag(Ks, 80, nsub * 16, ks, lane);
        s[nsub] = wmma16(a, b, s[nsub]);
      }
    }

    // online softmax (rows live on 16-lane halves; VGPR r -> row r+8*hi)
    const float scale = 0.125f;  // 1/sqrt(64)
#pragma unroll
    for (int r = 0; r < 8; ++r) {
      float v0 = s[0][r] * scale, v1 = s[1][r] * scale;
      float v2 = s[2][r] * scale, v3 = s[3][r] * scale;
      float mx = fmaxf(fmaxf(v0, v1), fmaxf(v2, v3));
#pragma unroll
      for (int off = 1; off < 16; off <<= 1)
        mx = fmaxf(mx, __shfl_xor(mx, off, 32));
      float mnew = fmaxf(mrow[r], mx);
      float alpha = __expf(mrow[r] - mnew);
      float p0 = __expf(v0 - mnew), p1 = __expf(v1 - mnew);
      float p2 = __expf(v2 - mnew), p3 = __expf(v3 - mnew);
      float rs = p0 + p1 + p2 + p3;
#pragma unroll
      for (int off = 1; off < 16; off <<= 1) rs += __shfl_xor(rs, off, 32);
      lrow[r] = lrow[r] * alpha + rs;
      mrow[r] = mnew;
      o[0][r] *= alpha; o[1][r] *= alpha; o[2][r] *= alpha; o[3][r] *= alpha;
      _Float16* pr = Ps + (wave * 16 + r + 8 * hi) * 80 + ln;
      pr[0] = (_Float16)p0; pr[16] = (_Float16)p1;
      pr[32] = (_Float16)p2; pr[48] = (_Float16)p3;
    }

    // O += P * V;  B[k=key][n=d] <- Vt[d][key]
#pragma unroll
    for (int ks = 0; ks < 64; ks += 32) {
      v16h a = load_a_frag(Ps, 80, wave * 16, ks, lane);
#pragma unroll
      for (int nsub = 0; nsub < 4; ++nsub) {
        v16h b = load_b_frag(Vt, 80, nsub * 16, ks, lane);
        o[nsub] = wmma16(a, b, o[nsub]);
      }
    }
    __syncthreads();
  }

  // ctx[b][q][h*64+d], f16
  const int b = bh >> 4, h = bh & 15;
#pragma unroll
  for (int r = 0; r < 8; ++r) {
    int q = qTile + wave * 16 + r + 8 * hi;
    float inv = 1.0f / lrow[r];
#pragma unroll
    for (int nsub = 0; nsub < 4; ++nsub) {
      int d = nsub * 16 + ln;
      ctx[((size_t)(b * 2048 + q)) * 1024 + h * 64 + d] =
          (_Float16)(o[nsub][r] * inv);
    }
  }
}

// ---------------------------------------------------------------------------
// Kernel 3: output projection. out[m,f] = sum_e ctx[m,e]*out_w[f,e] + out_b[f]
// Double-buffered; f16 A tile goes through async global->LDS, fp32 B tile via
// register convert path.
// ---------------------------------------------------------------------------
__global__ __launch_bounds__(256) void out_gemm_kernel(
    const _Float16* __restrict__ ctx, const float* __restrict__ out_w,
    const float* __restrict__ out_b, float* __restrict__ out) {
  __shared__ alignas(16) _Float16 As[2][128 * 40];
  __shared__ alignas(16) _Float16 Bs[2][128 * 40];
  const int tid = threadIdx.x, lane = tid & 31, wave = tid >> 5;
  const int mTile = blockIdx.x * 128;
  const int nTile = blockIdx.y * 128;

  const v8f zero = {0.f, 0.f, 0.f, 0.f, 0.f, 0.f, 0.f, 0.f};
  v8f acc[8];
#pragma unroll
  for (int i = 0; i < 8; ++i) acc[i] = zero;

  float4 fb[4];
  auto issue_a_async = [&](int k0, int buf) {
#pragma unroll
    for (int i = 0; i < 2; ++i) {        // ctx tile (f16): 512 x 16B chunks
      int idx = tid + i * 256;
      int row = idx >> 2, cg = idx & 3;
      async_copy_b128(&As[buf][row * 40 + cg * 8],
                      ctx + (size_t)(mTile + row) * 1024 + k0 + cg * 8);
    }
  };
  auto load_b_regs = [&](int k0) {
#pragma unroll
    for (int i = 0; i < 4; ++i) {
      int idx = tid + i * 256;
      int row = idx >> 3, cg = idx & 7;
      fb[i] = *(const float4*)(out_w + (size_t)(nTile + row) * 1024 + k0 + cg * 4);
    }
  };
  auto store_b = [&](int buf) {
#pragma unroll
    for (int i = 0; i < 4; ++i) {
      int idx = tid + i * 256;
      int row = idx >> 3, cg = idx & 7;
      _Float16* e = &Bs[buf][row * 40 + cg * 4];
      e[0] = (_Float16)fb[i].x; e[1] = (_Float16)fb[i].y;
      e[2] = (_Float16)fb[i].z; e[3] = (_Float16)fb[i].w;
    }
  };

  issue_a_async(0, 0);
  load_b_regs(0);
  store_b(0);
  async_wait0();
  __syncthreads();

  for (int k0 = 0; k0 < 1024; k0 += 32) {
    const int cur = (k0 >> 5) & 1;
    const bool more = (k0 + 32) < 1024;
    if (more) { issue_a_async(k0 + 32, cur ^ 1); load_b_regs(k0 + 32); }
    v16h a = load_a_frag(As[cur], 40, wave * 16, 0, lane);
#pragma unroll
    for (int ns = 0; ns < 8; ++ns) {
      v16h b = load_b_frag(Bs[cur], 40, ns * 16, 0, lane);
      acc[ns] = wmma16(a, b, acc[ns]);
    }
    if (more) store_b(cur ^ 1);
    async_wait0();
    __syncthreads();
  }

  const int ln = lane & 15, hi = lane >> 4;
  const int m_base = mTile + wave * 16;
#pragma unroll
  for (int ns = 0; ns < 8; ++ns) {
    int f = nTile + ns * 16 + ln;
    float bias = out_b[f];
#pragma unroll
    for (int r = 0; r < 8; ++r) {
      int m = m_base + r + 8 * hi;
      out[(size_t)m * 1024 + f] = acc[ns][r] + bias;
    }
  }
}

// ---------------------------------------------------------------------------
extern "C" void kernel_launch(void* const* d_in, const int* in_sizes, int n_in,
                              void* d_out, int out_size, void* d_ws,
                              size_t ws_size, hipStream_t stream) {
  (void)in_sizes; (void)n_in; (void)out_size; (void)ws_size;
  const float* inp   = (const float*)d_in[0];
  const float* qkv_w = (const float*)d_in[1];
  const float* qkv_b = (const float*)d_in[2];
  const float* out_w = (const float*)d_in[3];
  const float* out_b = (const float*)d_in[4];
  float* out = (float*)d_out;

  const size_t perTensor = (size_t)4 * 16 * 2048 * 64;  // 8,388,608 halfs
  _Float16* Qh  = (_Float16*)d_ws;
  _Float16* Kh  = Qh + perTensor;
  _Float16* Vh  = Kh + perTensor;
  _Float16* ctx = Vh + perTensor;       // total 64 MB of workspace

  // 1) QKV projection: M=8192, N=3072 -> 64 x 24 tiles of 128x128
  qkv_gemm_kernel<<<dim3(64, 24), 256, 0, stream>>>(inp, qkv_w, qkv_b, Qh, Kh, Vh);
  // 2) attention: 64 (b,h) pairs x 16 q-tiles of 128 rows
  attn_kernel<<<dim3(64, 16), 256, 0, stream>>>(Qh, Kh, Vh, ctx);
  // 3) output projection: M=8192, N=1024 -> 64 x 8 tiles
  out_gemm_kernel<<<dim3(64, 8), 256, 0, stream>>>(ctx, out_w, out_b, out);
}
